// nGPT_858993459364
// MI455X (gfx1250) — compile-verified, matrix operand order checked by reading, not run on hardware
//
#include <hip/hip_runtime.h>
#include <hip/hip_bf16.h>
#include <math.h>

// Model dims (fixed per reference)
#define SLEN   1024
#define DMODEL 1024
#define NH     16
#define NG     4
#define NL     4
#define HDIM   64
#define DFF    4096
#define VOCAB  32000
#define GHD    (NG * HDIM)   // 256
#define RT_HD  8.0f          // sqrt(64)
#define RT_D   32.0f         // sqrt(1024)

typedef __bf16 bf16_t;
typedef __attribute__((ext_vector_type(16))) __bf16 v16bf;
typedef __attribute__((ext_vector_type(8)))  __bf16 v8bf;
typedef __attribute__((ext_vector_type(8)))  float  v8f;

// -----------------------------------------------------------------------------
// WMMA helpers
// -----------------------------------------------------------------------------
// 16-bit A/B fragment for V_WMMA_*_16X16X32: row (or column) = lane&15,
// K-halves per documented layout: lanes 0-15 hold K {0..7, 16..23},
// lanes 16-31 hold K {8..15, 24..31}. Two 16B loads per lane.
__device__ __forceinline__ v16bf load_frag(const bf16_t* tile, int ld, int lane) {
  const bf16_t* rp = tile + (size_t)(lane & 15) * ld + ((lane >> 4) << 3);
  v8bf lo = *(const v8bf*)rp;          // K 0..7   (or 8..15)
  v8bf hi = *(const v8bf*)(rp + 16);   // K 16..23 (or 24..31)
  return __builtin_shufflevector(lo, hi, 0,1,2,3,4,5,6,7,8,9,10,11,12,13,14,15);
}

__device__ __forceinline__ v8f wmma_bf16(v16bf a, v16bf b, v8f c) {
  // (neg_a, A, neg_b, B, c_mod, C, reuse_a, reuse_b)
  return __builtin_amdgcn_wmma_f32_16x16x32_bf16(false, a, false, b, (short)0, c,
                                                 false, false);
}

// Async Global->LDS copy (ASYNCcnt-tracked). LDS byte address is the low 32
// bits of the generic shared-pointer (AS3 segment offset).
__device__ __forceinline__ void async_copy_b128(void* lds, const void* gptr) {
  asm volatile("global_load_async_to_lds_b128 %0, %1, off"
               :: "v"((unsigned int)(size_t)lds), "v"(gptr) : "memory");
}
__device__ __forceinline__ void wait_asynccnt_3() {
  asm volatile("s_wait_asynccnt 0x3" ::: "memory");
}
__device__ __forceinline__ void wait_asynccnt_0() {
  asm volatile("s_wait_asynccnt 0x0" ::: "memory");
}

__device__ __forceinline__ float block_reduce(float v, float* sh, int do_max) {
  int t = threadIdx.x;
  sh[t] = v;
  __syncthreads();
  for (int s = blockDim.x >> 1; s > 0; s >>= 1) {
    if (t < s) sh[t] = do_max ? fmaxf(sh[t], sh[t + s]) : (sh[t] + sh[t + s]);
    __syncthreads();
  }
  float r = sh[0];
  __syncthreads();
  return r;
}

// -----------------------------------------------------------------------------
// GEMM: C[M,N] = A[M,K] * B[N,K]^T + bias[N]   (A,B bf16 row-major, C f32)
// Block = 256 threads = 8 waves, block tile 64x128, wave tile 32x32.
// Double-buffered async Global->LDS staging; WMMA consumes from LDS.
// -----------------------------------------------------------------------------
#define LDSLD 40   // 32 data elems + 8 pad (80B rows, 16B aligned)

__global__ __launch_bounds__(256) void k_gemm(
    const bf16_t* __restrict__ A, const bf16_t* __restrict__ B,
    const float* __restrict__ bias, float* __restrict__ C,
    int M, int N, int K, int lda, int ldb, int ldc) {
  __shared__ bf16_t shA[2][64 * LDSLD];    // 2 x 5.0 KB
  __shared__ bf16_t shB[2][128 * LDSLD];   // 2 x 10.0 KB
  int t    = threadIdx.x;
  int lane = t & 31;
  int w    = t >> 5;
  int mblk = blockIdx.y * 64;
  int nblk = blockIdx.x * 128;
  (void)M; (void)N;

  const bf16_t* Ab = A + (size_t)mblk * lda;
  const bf16_t* Bb = B + (size_t)nblk * ldb;

  // cooperative copy tasks: A = 64 rows x 4 chunks (1/thread),
  // B = 128 rows x 4 chunks (2/thread); 16B (8 bf16) per chunk.
  int arow = t >> 2, achk = (t & 3) * 8;
  int b1row = t >> 2;
  int b2row = 64 + (t >> 2);

  auto issue = [&](int ks) {
    int p = ks & 1;
    int k = ks * 32;
    async_copy_b128(&shA[p][arow * LDSLD + achk],
                    Ab + (size_t)arow * lda + k + achk);
    async_copy_b128(&shB[p][b1row * LDSLD + achk],
                    Bb + (size_t)b1row * ldb + k + achk);
    async_copy_b128(&shB[p][b2row * LDSLD + achk],
                    Bb + (size_t)b2row * ldb + k + achk);
  };

  int m0w = (w >> 2) * 32;   // wave tile origin within block
  int n0w = (w & 3) * 32;

  v8f acc00 = {}, acc01 = {}, acc10 = {}, acc11 = {};
  int nks = K / 32;
  issue(0);
  for (int ks = 0; ks < nks; ++ks) {
    int p = ks & 1;
    if (ks + 1 < nks) {
      issue(ks + 1);       // overlap next-stage copies with this stage's WMMAs
      wait_asynccnt_3();   // stage ks's 3 copies complete (<=3 outstanding)
    } else {
      wait_asynccnt_0();
    }
    __syncthreads();       // all waves' tiles visible in LDS
    v16bf a0 = load_frag(&shA[p][(m0w)      * LDSLD], LDSLD, lane);
    v16bf a1 = load_frag(&shA[p][(m0w + 16) * LDSLD], LDSLD, lane);
    v16bf b0 = load_frag(&shB[p][(n0w)      * LDSLD], LDSLD, lane);
    v16bf b1 = load_frag(&shB[p][(n0w + 16) * LDSLD], LDSLD, lane);
    acc00 = wmma_bf16(a0, b0, acc00);
    acc01 = wmma_bf16(a0, b1, acc01);
    acc10 = wmma_bf16(a1, b0, acc10);
    acc11 = wmma_bf16(a1, b1, acc11);
    __syncthreads();       // reads done before this buffer is overwritten
  }

  // C fragment layout: lanes 0-15 -> M = v, lanes 16-31 -> M = v+8; N = lane&15.
  int rh  = (lane >> 4) * 8;
  int col = lane & 15;
  #pragma unroll
  for (int v = 0; v < 8; ++v) {
    int mA = mblk + m0w + rh + v, mB = mblk + m0w + 16 + rh + v;
    int nA = nblk + n0w + col,    nB = nblk + n0w + 16 + col;
    float bA = bias ? bias[nA] : 0.0f;
    float bB = bias ? bias[nB] : 0.0f;
    C[(size_t)mA * ldc + nA] = acc00[v] + bA;
    C[(size_t)mA * ldc + nB] = acc01[v] + bB;
    C[(size_t)mB * ldc + nA] = acc10[v] + bA;
    C[(size_t)mB * ldc + nB] = acc11[v] + bB;
  }
}

// -----------------------------------------------------------------------------
// Flash attention: 1 wave per (16-query tile, head). Online softmax.
// Q:(S,NH*HD) K,V:(S,NG*HD) bf16; O:(S,NH*HD) bf16. eff^2 pre-folded into Q.
// -----------------------------------------------------------------------------
__global__ __launch_bounds__(32) void k_attn(
    const bf16_t* __restrict__ Q, const bf16_t* __restrict__ Kc,
    const bf16_t* __restrict__ Vc, bf16_t* __restrict__ O) {
  __shared__ bf16_t pshm[16 * 32];   // P tile staging for A-frag re-layout
  int lane = threadIdx.x;
  int q0   = blockIdx.x * 16;
  int h    = blockIdx.y;
  int g    = h / (NH / NG);
  const int ldq = NH * HDIM, ldk = NG * HDIM;
  const bf16_t* qb = Q  + (size_t)h * HDIM;
  const bf16_t* kb = Kc + (size_t)g * HDIM;
  const bf16_t* vb = Vc + (size_t)g * HDIM;

  v16bf qa0 = load_frag(qb + (size_t)q0 * ldq + 0,  ldq, lane);
  v16bf qa1 = load_frag(qb + (size_t)q0 * ldq + 32, ldq, lane);

  int rh  = (lane >> 4) * 8;
  int col = lane & 15;
  int kh  = lane >> 4;

  float mrun[8], lrun[8];
  #pragma unroll
  for (int v = 0; v < 8; ++v) { mrun[v] = -1e30f; lrun[v] = 0.0f; }
  v8f o0 = {}, o1 = {}, o2 = {}, o3 = {};

  for (int j = 0; j < q0 + 16; j += 32) {   // 32-key KV chunks (causal)
    v8f s0 = {}, s1 = {};
    s0 = wmma_bf16(qa0, load_frag(kb + (size_t)j * ldk + 0,  ldk, lane), s0);
    s0 = wmma_bf16(qa1, load_frag(kb + (size_t)j * ldk + 32, ldk, lane), s0);
    s1 = wmma_bf16(qa0, load_frag(kb + (size_t)(j + 16) * ldk + 0,  ldk, lane), s1);
    s1 = wmma_bf16(qa1, load_frag(kb + (size_t)(j + 16) * ldk + 32, ldk, lane), s1);

    float p0[8], p1[8], sc[8];
    #pragma unroll
    for (int v = 0; v < 8; ++v) {
      int rowg = q0 + rh + v;
      p0[v] = (j + col      <= rowg) ? s0[v] * RT_HD : -1e30f;
      p1[v] = (j + 16 + col <= rowg) ? s1[v] * RT_HD : -1e30f;
    }
    #pragma unroll
    for (int v = 0; v < 8; ++v) {
      float rm = fmaxf(p0[v], p1[v]);
      for (int off = 1; off < 16; off <<= 1) rm = fmaxf(rm, __shfl_xor(rm, off, 32));
      float newm = fmaxf(mrun[v], rm);
      float e0 = __expf(p0[v] - newm), e1 = __expf(p1[v] - newm);
      float srow = e0 + e1;
      for (int off = 1; off < 16; off <<= 1) srow += __shfl_xor(srow, off, 32);
      sc[v]   = __expf(mrun[v] - newm);
      lrun[v] = lrun[v] * sc[v] + srow;
      mrun[v] = newm;
      p0[v] = e0; p1[v] = e1;
    }
    #pragma unroll
    for (int v = 0; v < 8; ++v) { o0[v] *= sc[v]; o1[v] *= sc[v]; o2[v] *= sc[v]; o3[v] *= sc[v]; }

    // stage P (16x32) through LDS, reload in A-fragment layout
    __syncthreads();
    #pragma unroll
    for (int v = 0; v < 8; ++v) {
      pshm[(rh + v) * 32 + col]      = (bf16_t)p0[v];
      pshm[(rh + v) * 32 + col + 16] = (bf16_t)p1[v];
    }
    __syncthreads();
    v16bf pa = load_frag((const bf16_t*)pshm, 32, lane);

    // V fragments (B operand): column n = output dim, K = key index
    v16bf vf0 = {}, vf1 = {}, vf2 = {}, vf3 = {};
    #pragma unroll
    for (int i = 0; i < 16; ++i) {
      int kk = (i < 8) ? (kh * 8 + i) : (16 + kh * 8 + (i - 8));
      const bf16_t* vr = vb + (size_t)(j + kk) * ldk + col;
      vf0[i] = vr[0]; vf1[i] = vr[16]; vf2[i] = vr[32]; vf3[i] = vr[48];
    }
    o0 = wmma_bf16(pa, vf0, o0);
    o1 = wmma_bf16(pa, vf1, o1);
    o2 = wmma_bf16(pa, vf2, o2);
    o3 = wmma_bf16(pa, vf3, o3);
  }

  #pragma unroll
  for (int v = 0; v < 8; ++v) {
    float inv = 1.0f / lrun[v];
    size_t base = (size_t)(q0 + rh + v) * ldq + (size_t)h * HDIM + col;
    O[base]      = (bf16_t)(o0[v] * inv);
    O[base + 16] = (bf16_t)(o1[v] * inv);
    O[base + 32] = (bf16_t)(o2[v] * inv);
    O[base + 48] = (bf16_t)(o3[v] * inv);
  }
}

// -----------------------------------------------------------------------------
// RoPE + cosine-norm per (token, head). One wave per 64-dim head row;
// lane p owns rotation pair (p, p+32). If sqk != nullptr also multiply by
// eff^2 = (sqk*sqrt(HD))^2 (eff folded fully into Q; K left unit-normed).
// -----------------------------------------------------------------------------
__global__ __launch_bounds__(32) void k_rope_norm(
    float* __restrict__ x, bf16_t* __restrict__ xb,
    const float* __restrict__ sqk, int nh) {
  int s = blockIdx.x, hh = blockIdx.y, p = threadIdx.x;
  size_t base = ((size_t)s * nh + hh) * HDIM;
  float x1 = x[base + p], x2 = x[base + p + 32];
  float inv = __expf(-((float)(2 * p) / (float)HDIM) * 9.210340372f); // ln(1e4)
  float ang = (float)s * inv;
  float c = __cosf(ang), sn = __sinf(ang);
  float r1 = x1 * c - x2 * sn;
  float r2 = x2 * c + x1 * sn;
  float ss = r1 * r1 + r2 * r2;
  for (int off = 1; off < 32; off <<= 1) ss += __shfl_xor(ss, off, 32);
  float nrm = fmaxf(sqrtf(ss), 1e-6f);
  float o1 = r1 / nrm, o2 = r2 / nrm;
  if (sqk) {
    float e1 = sqk[hh * HDIM + p]      * RT_HD;
    float e2 = sqk[hh * HDIM + p + 32] * RT_HD;
    o1 *= e1 * e1; o2 *= e2 * e2;
  }
  x[base + p] = o1; x[base + p + 32] = o2;
  xb[base + p] = (bf16_t)o1; xb[base + p + 32] = (bf16_t)o2;
}

// h = cosnorm(h + (eig*rtD) * (cosnorm(y) - h)), per row; also emit bf16 h.
__global__ __launch_bounds__(256) void k_resnorm(
    float* __restrict__ h, bf16_t* __restrict__ hb,
    const float* __restrict__ y, const float* __restrict__ eig) {
  __shared__ float sh[256];
  int s = blockIdx.x, t = threadIdx.x;
  float* hr = h + (size_t)s * DMODEL;
  const float* yr = y + (size_t)s * DMODEL;
  float ss = 0.0f;
  for (int d = t; d < DMODEL; d += 256) ss += yr[d] * yr[d];
  float yn = fmaxf(sqrtf(block_reduce(ss, sh, 0)), 1e-6f);
  float tv[4]; float ss2 = 0.0f; int i = 0;
  for (int d = t; d < DMODEL; d += 256, ++i) {
    float a = eig[d] * RT_D;
    float val = hr[d] + a * (yr[d] / yn - hr[d]);
    tv[i] = val; ss2 += val * val;
  }
  float tn = fmaxf(sqrtf(block_reduce(ss2, sh, 0)), 1e-6f);
  i = 0;
  for (int d = t; d < DMODEL; d += 256, ++i) {
    float o = tv[i] / tn;
    hr[d] = o; hb[(size_t)s * DMODEL + d] = (bf16_t)o;
  }
}

__global__ void k_embed(const int* __restrict__ x, const float* __restrict__ emb,
                        float* __restrict__ h, bf16_t* __restrict__ hb, long long n) {
  for (long long i = (long long)blockIdx.x * blockDim.x + threadIdx.x; i < n;
       i += (long long)gridDim.x * blockDim.x) {
    int s = (int)(i / DMODEL), d = (int)(i % DMODEL);
    float v = emb[(size_t)x[s] * DMODEL + d];
    h[i] = v; hb[i] = (bf16_t)v;
  }
}

__global__ void k_cvt(const float* __restrict__ src, bf16_t* __restrict__ dst, long long n) {
  for (long long i = (long long)blockIdx.x * blockDim.x + threadIdx.x; i < n;
       i += (long long)gridDim.x * blockDim.x)
    dst[i] = (bf16_t)src[i];
}

__global__ void k_swiglu(const float* __restrict__ u, const float* __restrict__ g,
                         const float* __restrict__ su, const float* __restrict__ sv,
                         bf16_t* __restrict__ act, long long n) {
  for (long long i = (long long)blockIdx.x * blockDim.x + threadIdx.x; i < n;
       i += (long long)gridDim.x * blockDim.x) {
    int f = (int)(i % DFF);
    float uu = u[i] * su[f];
    float gg = g[i] * sv[f] * RT_D;
    float si = gg / (1.0f + __expf(-gg));
    act[i] = (bf16_t)(uu * si);
  }
}

// out[s,:] = softmax((out[s,:] + out_b) * (s_z * rtD))
__global__ __launch_bounds__(256) void k_softmax(
    float* __restrict__ out, const float* __restrict__ ob, const float* __restrict__ sz) {
  __shared__ float sh[256];
  int s = blockIdx.x, t = threadIdx.x;
  float* row = out + (size_t)s * VOCAB;
  float mx = -1e30f;
  for (int v = t; v < VOCAB; v += 256) {
    float val = (row[v] + ob[v]) * (sz[v] * RT_D);
    row[v] = val;
    mx = fmaxf(mx, val);
  }
  mx = block_reduce(mx, sh, 1);
  float sum = 0.0f;
  for (int v = t; v < VOCAB; v += 256) sum += __expf(row[v] - mx);
  sum = block_reduce(sum, sh, 0);
  float inv = 1.0f / sum;
  for (int v = t; v < VOCAB; v += 256) row[v] = __expf(row[v] - mx) * inv;
}

// -----------------------------------------------------------------------------
// Host orchestration
// -----------------------------------------------------------------------------
extern "C" void kernel_launch(void* const* d_in, const int* in_sizes, int n_in,
                              void* d_out, int out_size, void* d_ws, size_t ws_size,
                              hipStream_t stream) {
  (void)in_sizes; (void)n_in; (void)out_size; (void)ws_size;
  const int*   x     = (const int*)d_in[0];
  const float* emb   = (const float*)d_in[1];
  const float* qw    = (const float*)d_in[2];
  const float* qbias = (const float*)d_in[3];
  const float* kw    = (const float*)d_in[4];
  const float* kbias = (const float*)d_in[5];
  const float* vw    = (const float*)d_in[6];
  const float* vbias = (const float*)d_in[7];
  const float* sqk   = (const float*)d_in[8];
  const float* ow    = (const float*)d_in[9];
  const float* obias = (const float*)d_in[10];
  const float* w_in  = (const float*)d_in[11];
  const float* b_in  = (const float*)d_in[12];
  const float* w_g   = (const float*)d_in[13];
  const float* b_g   = (const float*)d_in[14];
  const float* w_out = (const float*)d_in[15];
  const float* b_out = (const float*)d_in[16];
  const float* s_u   = (const float*)d_in[17];
  const float* s_v   = (const float*)d_in[18];
  const float* eig_a = (const float*)d_in[19];
  const float* eig_m = (const float*)d_in[20];
  const float* outw  = (const float*)d_in[21];
  const float* outb  = (const float*)d_in[22];
  const float* s_z   = (const float*)d_in[23];
  float* out = (float*)d_out;

  // Workspace layout (~73 MB total)
  char* ws = (char*)d_ws;
  const size_t MB = 1ull << 20;
  float*  Hf  = (float*) (ws + 0);                    //  4 MB  h f32 (S,D)
  bf16_t* Hb  = (bf16_t*)(ws + 4 * MB);               //  2 MB  h bf16
  float*  Qf  = (float*) (ws + 6 * MB);               //  4 MB  q f32
  float*  Kf  = (float*) (ws + 10 * MB);              //  1 MB  k f32
  float*  Vf  = (float*) (ws + 11 * MB);              //  1 MB  v f32
  bf16_t* Qb  = (bf16_t*)(ws + 12 * MB);              //  2 MB
  bf16_t* Kb  = (bf16_t*)(ws + 14 * MB);              // .5 MB
  bf16_t* Vb  = (bf16_t*)(ws + 14 * MB + 512 * 1024); // .5 MB
  bf16_t* Ab  = (bf16_t*)(ws + 15 * MB);              //  2 MB  attn out bf16
  float*  Uf  = (float*) (ws + 17 * MB);              // 16 MB  (S,DFF)
  float*  Gf  = (float*) (ws + 33 * MB);              // 16 MB  (S,DFF) / (S,D) reuse
  bf16_t* ACT = (bf16_t*)(ws + 49 * MB);              //  8 MB  swiglu act bf16
  bf16_t* WB  = (bf16_t*)(ws + 57 * MB);              // 16 MB  bf16 weight staging

  auto cvt = [&](const float* src, bf16_t* dst, long long n) {
    k_cvt<<<(int)((n + 2047) / 2048), 256, 0, stream>>>(src, dst, n);
  };
  auto gemm = [&](const bf16_t* A, const bf16_t* B, const float* bias, float* C,
                  int M, int N, int K, int lda, int ldb, int ldc) {
    dim3 g(N / 128, M / 64);
    k_gemm<<<g, 256, 0, stream>>>(A, B, bias, C, M, N, K, lda, ldb, ldc);
  };

  { // embedding gather
    long long n = (long long)SLEN * DMODEL;
    k_embed<<<(int)((n + 2047) / 2048), 256, 0, stream>>>(x, emb, Hf, Hb, n);
  }

  for (int i = 0; i < NL; ++i) {
    // QKV projections
    cvt(qw + (size_t)i * DMODEL * DMODEL, WB, (long long)DMODEL * DMODEL);
    gemm(Hb, WB, qbias + (size_t)i * DMODEL, Qf, SLEN, DMODEL, DMODEL, DMODEL, DMODEL, DMODEL);
    cvt(kw + (size_t)i * GHD * DMODEL, WB, (long long)GHD * DMODEL);
    gemm(Hb, WB, kbias + (size_t)i * GHD, Kf, SLEN, GHD, DMODEL, DMODEL, DMODEL, GHD);
    cvt(vw + (size_t)i * GHD * DMODEL, WB, (long long)GHD * DMODEL);
    gemm(Hb, WB, vbias + (size_t)i * GHD, Vf, SLEN, GHD, DMODEL, DMODEL, DMODEL, GHD);
    // RoPE + cosine norm (+ eff^2 folded into Q)
    k_rope_norm<<<dim3(SLEN, NH), 32, 0, stream>>>(Qf, Qb, sqk + (size_t)i * NH * HDIM, NH);
    k_rope_norm<<<dim3(SLEN, NG), 32, 0, stream>>>(Kf, Kb, nullptr, NG);
    cvt(Vf, Vb, (long long)SLEN * GHD);
    // flash attention
    k_attn<<<dim3(SLEN / 16, NH), 32, 0, stream>>>(Qb, Kb, Vb, Ab);
    // output projection + residual/eigen norm
    cvt(ow + (size_t)i * DMODEL * DMODEL, WB, (long long)DMODEL * DMODEL);
    gemm(Ab, WB, obias + (size_t)i * DMODEL, Gf, SLEN, DMODEL, DMODEL, DMODEL, DMODEL, DMODEL);
    k_resnorm<<<SLEN, 256, 0, stream>>>(Hf, Hb, Gf, eig_a + (size_t)i * DMODEL);
    // SwiGLU FFN
    cvt(w_in + (size_t)i * DFF * DMODEL, WB, (long long)DFF * DMODEL);
    gemm(Hb, WB, b_in + (size_t)i * DFF, Uf, SLEN, DFF, DMODEL, DMODEL, DMODEL, DFF);
    cvt(w_g + (size_t)i * DFF * DMODEL, WB, (long long)DFF * DMODEL);
    gemm(Hb, WB, b_g + (size_t)i * DFF, Gf, SLEN, DFF, DMODEL, DMODEL, DMODEL, DFF);
    {
      long long n = (long long)SLEN * DFF;
      k_swiglu<<<(int)((n + 2047) / 2048), 256, 0, stream>>>(
          Uf, Gf, s_u + (size_t)i * DFF, s_v + (size_t)i * DFF, ACT, n);
    }
    cvt(w_out + (size_t)i * DMODEL * DFF, WB, (long long)DMODEL * DFF);
    gemm(ACT, WB, b_out + (size_t)i * DMODEL, Gf, SLEN, DMODEL, DFF, DFF, DFF, DMODEL);
    k_resnorm<<<SLEN, 256, 0, stream>>>(Hf, Hb, Gf, eig_m + (size_t)i * DMODEL);
  }

  // vocab projection in 6400-row chunks (6400 % 128 == 0), then softmax
  const int CH = 6400;
  for (int c = 0; c < VOCAB; c += CH) {
    cvt(outw + (size_t)c * DMODEL, WB, (long long)CH * DMODEL);
    gemm(Hb, WB, nullptr, out + c, SLEN, CH, DMODEL, DMODEL, DMODEL, VOCAB);
  }
  k_softmax<<<SLEN, 256, 0, stream>>>(out, outb, s_z);
}
